// DynAAM_90580860273215
// MI455X (gfx1250) — matrix operand order repaired and checked by reading, not compiled, not run on hardware
//
#include <hip/hip_runtime.h>
#include <hip/hip_bf16.h>

typedef _Float16 v16h __attribute__((ext_vector_type(16)));
typedef _Float16 v8h  __attribute__((ext_vector_type(8)));
typedef _Float16 v4h  __attribute__((ext_vector_type(4)));
typedef float    v8f  __attribute__((ext_vector_type(8)));

#define D_DIM  512
#define B_ROWS 512
#define C_CLS  50000
#define K_SUB  3
#define N_ROWS (C_CLS * K_SUB)     // 150000 sub-center rows
#define NBLK   (N_ROWS / 48)       // 3125 blocks of 48 columns (16 groups of 3)
#define SCALE_F 64.0f

#if defined(__has_builtin) && __has_builtin(__builtin_amdgcn_sched_barrier)
#define SCHED_FENCE() __builtin_amdgcn_sched_barrier(0)
#else
#define SCHED_FENCE()
#endif

// ---------------------------------------------------------------------------
// Kernel 1/2: L2-normalize rows of a (nrows x 512) fp32 matrix into f16.
// One wave32 per row; each lane owns 4 float4's (stride-32 for coalescing).
// ---------------------------------------------------------------------------
__global__ __launch_bounds__(256) void normalize_rows_f16(
    const float* __restrict__ in, _Float16* __restrict__ out, int nrows)
{
    const int row  = blockIdx.x * 8 + (threadIdx.x >> 5);
    const int lane = threadIdx.x & 31;
    if (row >= nrows) return;

    const float4* src = (const float4*)(in + (size_t)row * D_DIM);
    float4 v[4];
    float  ss = 0.0f;
#pragma unroll
    for (int i = 0; i < 4; ++i) {
        v[i] = src[lane + 32 * i];
        ss += v[i].x * v[i].x + v[i].y * v[i].y + v[i].z * v[i].z + v[i].w * v[i].w;
    }
#pragma unroll
    for (int off = 16; off > 0; off >>= 1) ss += __shfl_xor(ss, off, 32);

    const float inv = 1.0f / fmaxf(sqrtf(ss), 1e-12f);

    v4h* dst = (v4h*)(out + (size_t)row * D_DIM);
#pragma unroll
    for (int i = 0; i < 4; ++i) {
        v4h h;
        h[0] = (_Float16)(v[i].x * inv);
        h[1] = (_Float16)(v[i].y * inv);
        h[2] = (_Float16)(v[i].z * inv);
        h[3] = (_Float16)(v[i].w * inv);
        dst[lane + 32 * i] = h;
    }
}

// ---------------------------------------------------------------------------
// Kernel 3: cos = Xn (512x512) @ Wn^T (512x150000) with fused max-over-3.
// Block = 8 waves. Block owns M-tile (16 rows, staged in LDS); wave w owns
// n-block nb covering Wn rows [48*nb, 48*nb+48) = output groups [16*nb,16*nb+16).
// Distance-2 software pipeline: 3 rotating fragment slots; sched_barriers pin
// the [loads(k+2)] | [wmma(k)] interleave so WMMAs never wait on fresh loads.
// ---------------------------------------------------------------------------
__global__ __launch_bounds__(256) void gemm_maxk_kernel(
    const _Float16* __restrict__ Xn,
    const _Float16* __restrict__ Wn,
    float* __restrict__ out)
{
    __shared__ _Float16 As[16 * D_DIM];     // 16 KB: A tile, row-major
    __shared__ float    red[8 * 16 * 48];   // 24 KB: per-wave epilogue pad

    const int wave  = threadIdx.x >> 5;
    const int lane  = threadIdx.x & 31;
    const int mbase = blockIdx.y * 16;

    // Cooperative stage of the 16x512 f16 A tile (16 KB = 1024 uint4).
    {
        const uint4* src = (const uint4*)(Xn + (size_t)mbase * D_DIM);
        uint4*       dst = (uint4*)As;
        for (int i = threadIdx.x; i < 1024; i += 256) dst[i] = src[i];
    }
    __syncthreads();

    const int nb = blockIdx.x * 8 + wave;
    if (nb >= NBLK) return;

    const int r  = lane & 15;   // A row / B column within tile
    const int kh = lane >> 4;   // K-half select

    // B per-lane base pointers: Wn row (= B column), 16 contiguous K-halfs.
    const _Float16* bp0  = Wn + ((size_t)(nb * 48 +  0 + r)) * D_DIM + kh * 16;
    const _Float16* bp1  = Wn + ((size_t)(nb * 48 + 16 + r)) * D_DIM + kh * 16;
    const _Float16* bp2  = Wn + ((size_t)(nb * 48 + 32 + r)) * D_DIM + kh * 16;
    // A fragment: lo chunk at arow+k0, hi chunk at arow+k0+8 (16*kh folded in).
    const _Float16* arow = As + r * D_DIM + 16 * kh;

    v8f acc0 = {}, acc1 = {}, acc2 = {};

    // Rotating fragment slots (distance-2 pipeline).
    v16h a[3], b0[3], b1[3], b2[3];

#define LOAD_STAGE(slot, k0)                                                   \
    do {                                                                       \
        v8h alo = *(const v8h*)(arow + (k0));                                  \
        v8h ahi = *(const v8h*)(arow + (k0) + 8);                              \
        a[slot]  = __builtin_shufflevector(alo, ahi,                           \
                                           0, 1, 2, 3, 4, 5, 6, 7,             \
                                           8, 9, 10, 11, 12, 13, 14, 15);      \
        b0[slot] = *(const v16h*)(bp0 + (k0));                                 \
        b1[slot] = *(const v16h*)(bp1 + (k0));                                 \
        b2[slot] = *(const v16h*)(bp2 + (k0));                                 \
    } while (0)

    LOAD_STAGE(0, 0);
    LOAD_STAGE(1, 32);
    SCHED_FENCE();

#pragma unroll
    for (int ks = 0; ks < 16; ++ks) {
        const int cur = ks % 3;
        const int nxt = (ks + 2) % 3;
        if (ks < 14) LOAD_STAGE(nxt, (ks + 2) * 32);
        SCHED_FENCE();   // keep prefetch loads issued before this stage's WMMAs
        acc0 = __builtin_amdgcn_wmma_f32_16x16x32_f16(false, a[cur], false, b0[cur],
                                                      (short)0, acc0, false, false);
        acc1 = __builtin_amdgcn_wmma_f32_16x16x32_f16(false, a[cur], false, b1[cur],
                                                      (short)0, acc1, false, false);
        acc2 = __builtin_amdgcn_wmma_f32_16x16x32_f16(false, a[cur], false, b2[cur],
                                                      (short)0, acc2, false, false);
        SCHED_FENCE();   // keep next stage's loads below this stage's WMMAs
    }
#undef LOAD_STAGE

    // Epilogue: spill 16x48 to per-wave LDS, reduce each group of 3 columns.
    float* S = red + wave * (16 * 48);
#pragma unroll
    for (int v = 0; v < 8; ++v) {
        const int row = v + 8 * kh;        // C/D layout: lane>=16 holds rows 8..15
        S[row * 48 +  0 + r] = acc0[v];
        S[row * 48 + 16 + r] = acc1[v];
        S[row * 48 + 32 + r] = acc2[v];
    }
    // Same-wave LDS RAW; compiler inserts s_wait_dscnt.
    const size_t gcol = (size_t)nb * 16 + r;   // lane r -> output group column
#pragma unroll
    for (int v = 0; v < 8; ++v) {
        const int row = v + 8 * kh;
        const float m0 = S[row * 48 + 3 * r + 0];
        const float m1 = S[row * 48 + 3 * r + 1];
        const float m2 = S[row * 48 + 3 * r + 2];
        const float m  = SCALE_F * fmaxf(m0, fmaxf(m1, m2));
        // Non-temporal: keep the 102MB output stream out of L2 (weights live there).
        __builtin_nontemporal_store(m, out + (size_t)(mbase + row) * C_CLS + gcol);
    }
}

// ---------------------------------------------------------------------------
// Kernel 4: per-row label fixup. Recompute the 3 sub-center cosines from the
// same f16 normalized operands, apply the dynamic margin, overwrite 1 element.
// ---------------------------------------------------------------------------
__global__ __launch_bounds__(256) void fixup_labels(
    const _Float16* __restrict__ Xn, const _Float16* __restrict__ Wn,
    const int* __restrict__ label, const float* __restrict__ margins,
    float* __restrict__ out)
{
    const int b    = blockIdx.x * 8 + (threadIdx.x >> 5);
    const int lane = threadIdx.x & 31;
    if (b >= B_ROWS) return;

    const int lab = label[b];
    const v8h* xv = (const v8h*)(Xn + (size_t)b * D_DIM);
    const v8h  x0 = xv[lane * 2 + 0];
    const v8h  x1 = xv[lane * 2 + 1];

    float c[3];
#pragma unroll
    for (int j = 0; j < 3; ++j) {
        const v8h* wv = (const v8h*)(Wn + ((size_t)lab * K_SUB + j) * D_DIM);
        const v8h  w0 = wv[lane * 2 + 0];
        const v8h  w1 = wv[lane * 2 + 1];
        float acc = 0.0f;
#pragma unroll
        for (int i = 0; i < 8; ++i)
            acc += (float)x0[i] * (float)w0[i] + (float)x1[i] * (float)w1[i];
#pragma unroll
        for (int off = 16; off > 0; off >>= 1) acc += __shfl_xor(acc, off, 32);
        c[j] = acc;
    }

    if (lane == 0) {
        const float target = fmaxf(c[0], fmaxf(c[1], c[2]));
        const float shift  = 0.5f * (fabsf(c[0] - c[1]) + fabsf(c[0] - c[2]));
        const float wm     = margins[lab] * (1.0f + shift);
        out[(size_t)b * C_CLS + lab] = SCALE_F * (target - wm);
    }
}

// ---------------------------------------------------------------------------
extern "C" void kernel_launch(void* const* d_in, const int* in_sizes, int n_in,
                              void* d_out, int out_size, void* d_ws, size_t ws_size,
                              hipStream_t stream)
{
    const float* input   = (const float*)d_in[0];   // (512, 512) fp32
    const int*   label   = (const int*)d_in[1];     // (512,) int
    const float* weights = (const float*)d_in[2];   // (150000, 512) fp32
    const float* margins = (const float*)d_in[3];   // (50000,) fp32
    float* out = (float*)d_out;                     // (512, 50000) fp32

    // Workspace: Xn f16 (512 KB) then Wn f16 (153.6 MB).
    _Float16* Xn = (_Float16*)d_ws;
    _Float16* Wn = (_Float16*)((char*)d_ws + (size_t)B_ROWS * D_DIM * sizeof(_Float16));

    // 1) normalize input rows -> f16
    normalize_rows_f16<<<dim3(B_ROWS / 8), dim3(256), 0, stream>>>(input, Xn, B_ROWS);
    // 2) normalize weight rows -> f16
    normalize_rows_f16<<<dim3(N_ROWS / 8), dim3(256), 0, stream>>>(weights, Wn, N_ROWS);
    // 3) WMMA GEMM + fused max-over-subcenters (grid: 391 n-superblocks x 32 m-tiles)
    gemm_maxk_kernel<<<dim3((NBLK + 7) / 8, B_ROWS / 16), dim3(256), 0, stream>>>(Xn, Wn, out);
    // 4) dynamic-margin fixup at label positions
    fixup_labels<<<dim3(B_ROWS / 8), dim3(256), 0, stream>>>(Xn, Wn, label, margins, out);
}
